// SampleConditioningNetwork16_45251775431070
// MI455X (gfx1250) — matrix operand order repaired and checked by reading, not compiled, not run on hardware
//
#include <hip/hip_runtime.h>
#include <hip/hip_bf16.h>

// ---------------------------------------------------------------------------
// WaveNet fast-generation on gfx1250 (MI455X).
//
// Time-aligned recurrences (k=2 dilated convs, dilations 1/2/4/8):
//   h1_t = W1_0 x_{t-1}  + W1_1 x_t        (active t >= 1)
//   h2_t = W2_0 h1_{t-2} + W2_1 h1_t       (active t >= 3)
//   h3_t = W3_0 h2_{t-4} + W3_1 h2_t       (active t >= 7)
//   x_{t+1} = W4_0 h3_{t-8} + W4_1 h3_t    (active t >= 15, 15 samples)
// Final output window = x_15 .. x_30.
//
// One workgroup (8 wave32s) owns 32 batch rows; ALL ring state lives in the
// 320KB CDNA5 LDS. Matmuls are v_wmma_f32_16x16x32_bf16 (f32 accumulate).
// ---------------------------------------------------------------------------

typedef __attribute__((ext_vector_type(16))) __bf16 v16bf;
typedef __attribute__((ext_vector_type(8)))  __bf16 v8bf;
typedef __attribute__((ext_vector_type(8)))  float  v8f;

constexpr int NC  = 256;          // channels
constexpr int RB  = 32;           // batch rows per workgroup
constexpr int ENT = RB * NC;      // bf16 elements per ring entry (16 KB)
// LDS: X ring(2) + H1 ring(3) + H2 ring(5) + H3 ring(9) = 19 entries = 304 KB
constexpr size_t SHMEM_BYTES = (size_t)19 * ENT * sizeof(__bf16);
// Packed weight buffer: 4 layers * 2 taps * 16 ntile * 8 ktile * 32 lane * 16 bf16
constexpr size_t WPL        = (size_t)2 * 16 * 8 * 32 * 16;   // elems per layer
constexpr size_t PACK_ELEMS = 4 * WPL;                        // 524288 elems = 1 MB

__device__ __forceinline__ v8f wmma_bf16(v16bf a, v16bf b, v8f c) {
  // (neg_a, A, neg_b, B, c_mod, C, reuse_a, reuse_b)
  return __builtin_amdgcn_wmma_f32_16x16x32_bf16(false, a, false, b,
                                                 (short)0, c, false, false);
}

// A fragment (16x32 bf16) from an LDS ring entry laid out [RB rows][256 ch].
// Layout: lanes 0-15 (M=lane): elems 0..15 = K 0..15;
//         lanes 16-31 (M=lane-16): K 16..31 (matches load_b).
__device__ __forceinline__ v16bf load_a(const __bf16* __restrict__ e,
                                        int at, int kt, int lane) {
  const int m  = at * 16 + (lane & 15);
  const int k0 = kt * 32 + (lane >> 4) * 16;
  const __bf16* p = e + m * NC + k0;
  union { v16bf v; v8bf h[2]; } u;
  u.h[0] = *(const v8bf*)(p);        // K k0..k0+7      -> ds_load_b128
  u.h[1] = *(const v8bf*)(p + 8);    // K k0+8..k0+15   -> ds_load_b128
  return u.v;
}

// B fragment (32x16 bf16). Contiguous-k layout (as load_a): lane n in 0-15 holds K=0..15 (elem e
// <-> K = e), lanes 16-31 hold K=16..31. So lane elem e <-> K = kt*32 +
// (lane>>4)*16 + e, N = nt*16 + (lane&15).  B[k][n] = W[n][k] = w[n][k][tap].
template <bool PACKED>
__device__ __forceinline__ v16bf load_b(const float* __restrict__ wraw,
                                        const __bf16* __restrict__ wpackL,
                                        int tap, int nt, int kt, int lane) {
  if constexpr (PACKED) {
    const __bf16* p =
        wpackL + ((((size_t)tap * 16 + nt) * 8 + kt) * 32 + lane) * 16;
    return *(const v16bf*)p;         // 32B/lane -> 2x global_load_b128
  } else {
    const int n  = nt * 16 + (lane & 15);
    const int kb = kt * 32 + (lane >> 4) * 16;
    v16bf b;
#pragma unroll
    for (int e = 0; e < 16; ++e)
      b[e] = (__bf16)wraw[((size_t)n * NC + kb + e) * 2 + tap];
    return b;
  }
}

// One conv layer for this WG's 32 rows: out[RBx256] = inPrev*W0^T + inCur*W1^T.
// Wave w owns N-tiles {2w, 2w+1}; B frags held in VGPRs across both M-tiles.
template <bool PACKED, bool FINAL>
__device__ __forceinline__ void do_layer(const __bf16* __restrict__ inPrev,
                                         const __bf16* __restrict__ inCur,
                                         const float* __restrict__ wraw,
                                         const __bf16* __restrict__ wpackL,
                                         __bf16* __restrict__ outEnt,
                                         float* __restrict__ gout, int goutCol,
                                         int row0, int lane, int wave) {
  const int laneLo = lane & 15;
  const int laneHi = lane >> 4;
#pragma unroll
  for (int ni = 0; ni < 2; ++ni) {
    const int nt = wave * 2 + ni;
    v8f acc0 = {};
    v8f acc1 = {};
#pragma unroll
    for (int kt = 0; kt < 8; ++kt) {
      v16bf b0 = load_b<PACKED>(wraw, wpackL, 0, nt, kt, lane);
      v16bf b1 = load_b<PACKED>(wraw, wpackL, 1, nt, kt, lane);
      v16bf a0p = load_a(inPrev, 0, kt, lane);
      v16bf a0c = load_a(inCur, 0, kt, lane);
      acc0 = wmma_bf16(a0p, b0, acc0);
      acc0 = wmma_bf16(a0c, b1, acc0);
      v16bf a1p = load_a(inPrev, 1, kt, lane);
      v16bf a1c = load_a(inCur, 1, kt, lane);
      acc1 = wmma_bf16(a1p, b0, acc1);
      acc1 = wmma_bf16(a1c, b1, acc1);
    }
    // C/D layout: VGPR v, lanes 0-15 -> M=v, N=lane; lanes 16-31 -> M=8+v.
    const int n = nt * 16 + laneLo;
#pragma unroll
    for (int at = 0; at < 2; ++at) {
      v8f acc = at ? acc1 : acc0;
      const int mb = at * 16 + laneHi * 8;
#pragma unroll
      for (int v = 0; v < 8; ++v) {
        const float f = acc[v];
        const int m = mb + v;
        outEnt[m * NC + n] = (__bf16)f;
        if constexpr (FINAL)
          gout[((size_t)(row0 + m) * NC + n) * 16 + goutCol] = f;
      }
    }
  }
}

// Pack raw f32 OIH weights into WMMA-lane-ordered bf16 fragments in d_ws.
__global__ void pack_weights_kernel(const float* __restrict__ w1,
                                    const float* __restrict__ w2,
                                    const float* __restrict__ w3,
                                    const float* __restrict__ w4,
                                    __bf16* __restrict__ dst) {
  const int g = blockIdx.x * 256 + threadIdx.x;  // 8*16*8*32 = 32768 lanes
  if (g >= 8 * 16 * 8 * 32) return;
  const int lane = g & 31;
  const int kt = (g >> 5) & 7;
  const int nt = (g >> 8) & 15;
  const int lt = g >> 12;  // layer*2 + tap
  const int layer = lt >> 1, tap = lt & 1;
  const float* w = (layer == 0) ? w1 : (layer == 1) ? w2 : (layer == 2) ? w3 : w4;
  const int n  = nt * 16 + (lane & 15);
  const int kb = kt * 32 + (lane >> 4) * 16;
  __bf16* o = dst + (size_t)g * 16;
#pragma unroll
  for (int e = 0; e < 16; ++e)
    o[e] = (__bf16)w[((size_t)n * NC + kb + e) * 2 + tap];
}

template <bool PACKED>
__global__ __launch_bounds__(256, 1) void wavenet_recurrence_kernel(
    const float* __restrict__ x, const float* __restrict__ w1,
    const float* __restrict__ w2, const float* __restrict__ w3,
    const float* __restrict__ w4, const __bf16* __restrict__ wsB,
    float* __restrict__ out) {
  extern __shared__ __bf16 lds[];
  __bf16* Xr = lds;             // 2 entries: x_{t-1}, x_t
  __bf16* H1 = lds + 2 * ENT;   // 3-entry ring
  __bf16* H2 = lds + 5 * ENT;   // 5-entry ring
  __bf16* H3 = lds + 10 * ENT;  // 9-entry ring

  const int tid = threadIdx.x;
  const int lane = tid & 31;
  const int wave = tid >> 5;
  const int row0 = blockIdx.x * RB;

  const __bf16* wp1 = PACKED ? wsB + 0 * WPL : nullptr;
  const __bf16* wp2 = PACKED ? wsB + 1 * WPL : nullptr;
  const __bf16* wp3 = PACKED ? wsB + 2 * WPL : nullptr;
  const __bf16* wp4 = PACKED ? wsB + 3 * WPL : nullptr;

  // Stage x_0 (bf16) into X ring slot 0.
  for (int i = tid; i < RB * NC; i += 256) {
    const int r = i >> 8, c = i & 255;
    Xr[i] = (__bf16)x[((size_t)(row0 + r) * NC + c) * 16 + 0];
  }
  __syncthreads();

#pragma unroll 1
  for (int t = 1; t <= 29; ++t) {
    if (t <= 15) {  // stage x_t from input; t>15 samples come from layer 4
      for (int i = tid; i < RB * NC; i += 256) {
        const int r = i >> 8, c = i & 255;
        const float v = x[((size_t)(row0 + r) * NC + c) * 16 + t];
        Xr[(t & 1) * ENT + i] = (__bf16)v;
        if (t == 15)  // out[:, :, 0] = x[:, :, 15]
          out[((size_t)(row0 + r) * NC + c) * 16 + 0] = v;
      }
    }
    __syncthreads();

    do_layer<PACKED, false>(Xr + ((t - 1) & 1) * ENT, Xr + (t & 1) * ENT, w1,
                            wp1, H1 + (t % 3) * ENT, nullptr, 0, row0, lane,
                            wave);
    __syncthreads();
    if (t >= 3) {
      do_layer<PACKED, false>(H1 + ((t - 2) % 3) * ENT, H1 + (t % 3) * ENT, w2,
                              wp2, H2 + (t % 5) * ENT, nullptr, 0, row0, lane,
                              wave);
      __syncthreads();
    }
    if (t >= 7) {
      do_layer<PACKED, false>(H2 + ((t - 4) % 5) * ENT, H2 + (t % 5) * ENT, w3,
                              wp3, H3 + (t % 9) * ENT, nullptr, 0, row0, lane,
                              wave);
      __syncthreads();
    }
    if (t >= 15) {  // y_t -> new sample x_{t+1}: bf16 to X ring, f32 to out
      do_layer<PACKED, true>(H3 + ((t - 8) % 9) * ENT, H3 + (t % 9) * ENT, w4,
                             wp4, Xr + ((t + 1) & 1) * ENT, out, t - 14, row0,
                             lane, wave);
      __syncthreads();
    }
  }
}

extern "C" void kernel_launch(void* const* d_in, const int* in_sizes, int n_in,
                              void* d_out, int out_size, void* d_ws,
                              size_t ws_size, hipStream_t stream) {
  const float* x  = (const float*)d_in[0];
  const float* w1 = (const float*)d_in[1];
  const float* w2 = (const float*)d_in[2];
  const float* w3 = (const float*)d_in[3];
  const float* w4 = (const float*)d_in[4];
  float* out = (float*)d_out;

  const int nblocks = 8192 / RB;  // 256 workgroups, 32 batch rows each
  const bool packed = (d_ws != nullptr) &&
                      (ws_size >= PACK_ELEMS * sizeof(__bf16));
  if (packed) {
    pack_weights_kernel<<<(8 * 16 * 8 * 32) / 256, 256, 0, stream>>>(
        w1, w2, w3, w4, (__bf16*)d_ws);
    wavenet_recurrence_kernel<true>
        <<<nblocks, 256, SHMEM_BYTES, stream>>>(x, w1, w2, w3, w4,
                                                (const __bf16*)d_ws, out);
  } else {
    wavenet_recurrence_kernel<false>
        <<<nblocks, 256, SHMEM_BYTES, stream>>>(x, w1, w2, w3, w4, nullptr,
                                                out);
  }
}
